// AdaptivePRISM_274877907071
// MI455X (gfx1250) — matrix operand order repaired and checked by reading, not compiled, not run on hardware
//
#include <hip/hip_runtime.h>
#include <cstddef>
#include <cstdint>

// ---------------- problem constants (match reference) ----------------
#define UCNT   80000
#define ICNT   40000
#define DIM    128
#define NNODE  120000        // UCNT + ICNT
#define E2CNT  1000000       // 2*E directed edges
#define TIN    384
#define VIN    4096
#define HID1   256

typedef __attribute__((ext_vector_type(16))) __bf16 v16bf;
typedef __attribute__((ext_vector_type(8)))  float  v8f;
typedef __attribute__((ext_vector_type(4))) unsigned int u32x4;
typedef __attribute__((ext_vector_type(8))) int          i32x8;
typedef __attribute__((ext_vector_type(4))) int          i32x4;

// ---------------- elementwise utils ----------------
__global__ void k_zero(float* __restrict__ p, long long n){
  long long i = (long long)blockIdx.x*blockDim.x + threadIdx.x;
  if (i < n) p[i] = 0.0f;
}
__global__ void k_copy(float* __restrict__ d, const float* __restrict__ s, long long n){
  long long i = (long long)blockIdx.x*blockDim.x + threadIdx.x;
  if (i < n) d[i] = s[i];
}

// W[K][Nc] row-major f32  ->  Bt[Nc][K] bf16 (RNE via clang f32->bf16 conversion)
__global__ void k_conv_bt(const float* __restrict__ W, __bf16* __restrict__ Bt, int K, int Nc){
  long long i = (long long)blockIdx.x*blockDim.x + threadIdx.x;
  if (i >= (long long)K*Nc) return;
  int k = (int)(i / Nc), n = (int)(i % Nc);
  Bt[(size_t)n*K + k] = (__bf16)W[i];
}

// ---------------- WMMA GEMM with TDM-staged B panel ----------------
// C[M,Nc] = epi(A[M,K] @ Bt[Nc,K]^T + bias)
// mode 1: BN + PReLU   mode 2: ReLU
// Block = 128 threads = 4 waves; each wave owns one 16x16 C tile; block covers 64 cols.
// B panel (64 cols x 128 K bf16) is DMA'd into LDS by the Tensor Data Mover each K-chunk,
// with per-row padding (256B data + 32B pad) so ds_load_b128 fragment reads stay ~conflict-free.
// grid = (M/16, Nc/64). Requires K % 128 == 0 (384, 4096, 256, 128 all qualify).
#define KC    128
#define BROW  144            // 128 data + 16 pad bf16 elements per LDS row (288B, 32B-aligned)

__global__ void __launch_bounds__(128)
k_gemm_bf16(const float* __restrict__ A, int K,
            const __bf16* __restrict__ Bt,
            float* __restrict__ C, int Nc,
            const float* __restrict__ bias,
            const float* __restrict__ g, const float* __restrict__ bt,
            const float* __restrict__ aP, int mode)
{
  __shared__ __align__(128) __bf16 ldsB[64 * BROW];   // 18.4 KB
  const int lane = threadIdx.x & 31;
  const int wave = threadIdx.x >> 5;
  const int tr = blockIdx.x;
  const int hi = lane >> 4;                 // half-wave select
  const int r  = tr*16 + (lane & 15);       // A row this lane feeds
  const int c0 = blockIdx.y*64;             // block column base
  const int cl = wave*16 + (lane & 15);     // column within block == LDS row
  const int c  = c0 + cl;                   // C column this lane owns
  const float* aRow = A + (size_t)r*K;
  v8f acc = {};

  const unsigned long long gbase = (unsigned long long)(uintptr_t)(Bt + (size_t)c0*K);
  const unsigned lds_base = (unsigned)(uintptr_t)&ldsB[0];   // LDS aperture: low 32 bits = LDS offset

  for (int k0 = 0; k0 < K; k0 += KC){
    __syncthreads();                        // previous chunk fully consumed
    if (threadIdx.x < 32){                  // one wave drives the TDM (EXEC ignored by TDM)
      unsigned long long ga = gbase + (unsigned long long)k0*2ull;
      u32x4 g0; i32x8 g1; i32x4 g2; i32x4 g3; i32x8 gx;
      // D# group0: count=1 | lds_addr | global_addr | type=2 ("image")
      g0[0] = 1u;
      g0[1] = lds_base;
      g0[2] = (unsigned)(ga & 0xFFFFFFFFull);
      g0[3] = (unsigned)((ga >> 32) & 0x01FFFFFFull) | (2u << 30);
      // D# group1: data_size=2B | pad_enable, interval 64 DWORDs (code5), amount 8 DWORDs (code7)
      g1[0] = (1 << 16) | (1 << 20) | (5 << 22) | (7 << 25);
      g1[1] = (int)((unsigned)K << 16);                                   // tensor_dim0 lo16
      g1[2] = (int)(((unsigned)K >> 16) & 0xFFFFu) | (int)((unsigned)Nc << 16); // dim0 hi | dim1 lo
      g1[3] = (int)(((unsigned)Nc >> 16) & 0xFFFFu) | (KC << 16);         // dim1 hi | tile_dim0=128
      g1[4] = 64;                                                         // tile_dim1=64, tile_dim2=0
      g1[5] = K;                                                          // tensor_dim0_stride lo32
      g1[6] = 0;                                                          // stride hi | dim1_stride lo
      g1[7] = 0;
      g2[0]=0; g2[1]=0; g2[2]=0; g2[3]=0;                                 // 2-D: dims 2+ unused
      g3[0]=0; g3[1]=0; g3[2]=0; g3[3]=0;
      gx[0]=0; gx[1]=0; gx[2]=0; gx[3]=0; gx[4]=0; gx[5]=0; gx[6]=0; gx[7]=0;
      __builtin_amdgcn_tensor_load_to_lds(g0, g1, g2, g3, gx, 0);         // 6-arg toolchain form
      __builtin_amdgcn_s_wait_tensorcnt(0);
    }
    __syncthreads();                        // tile visible to all waves

    const __bf16* bRow = &ldsB[cl * BROW];
    for (int kk = 0; kk < KC; kk += 32){
      const int kg = k0 + kk;
      if (kg + 64 <= K) __builtin_prefetch(aRow + kg + 32, 0, 1);  // global_prefetch_b8 on A stream
      // A k order used here (B matches): lanes0-15 hold K 0..15, lanes16-31 hold K 16..31
      const float4* ap  = reinterpret_cast<const float4*>(aRow + kg + hi*16);
      const float4* ap2 = reinterpret_cast<const float4*>(aRow + kg + hi*16 + 8);
      float4 x0 = ap[0], x1 = ap[1], x2 = ap2[0], x3 = ap2[1];
      // B fragment: lane's column row in LDS, 16 consecutive K (lo half 0..15, hi half 16..31)
      v16bf bv = *reinterpret_cast<const v16bf*>(bRow + kk + hi*16);
      v16bf av;
      av[0]=(__bf16)x0.x; av[1]=(__bf16)x0.y; av[2]=(__bf16)x0.z; av[3]=(__bf16)x0.w;
      av[4]=(__bf16)x1.x; av[5]=(__bf16)x1.y; av[6]=(__bf16)x1.z; av[7]=(__bf16)x1.w;
      av[8]=(__bf16)x2.x; av[9]=(__bf16)x2.y; av[10]=(__bf16)x2.z; av[11]=(__bf16)x2.w;
      av[12]=(__bf16)x3.x; av[13]=(__bf16)x3.y; av[14]=(__bf16)x3.z; av[15]=(__bf16)x3.w;
      acc = __builtin_amdgcn_wmma_f32_16x16x32_bf16(false, av, false, bv, (short)0, acc,
                                                    false, false);
    }
  }
  const float bi = bias[c];
  const float pa = (mode==1) ? aP[0] : 0.0f;
  const float sc = (mode==1) ? g[c]*rsqrtf(1.0f + 1e-5f) : 0.0f;
  const float sb = (mode==1) ? bt[c] : 0.0f;
  #pragma unroll
  for (int j = 0; j < 8; j++){
    int row = tr*16 + hi*8 + j;             // C layout: VGPR j -> M=j (+8 for hi half)
    float v = acc[j] + bi;
    if (mode == 1){ v = v*sc + sb; v = (v >= 0.0f) ? v : pa*v; }
    else if (mode == 2){ v = fmaxf(v, 0.0f); }
    C[(size_t)row*Nc + c] = v;
  }
}

// ---------------- graph kernels ----------------
__global__ void k_deg(const int* __restrict__ dst, float* __restrict__ deg, int ne){
  int e = blockIdx.x*blockDim.x + threadIdx.x;
  if (e < ne) atomicAdd(&deg[dst[e]], 1.0f);
}
__global__ void k_dinv(const float* __restrict__ deg, float* __restrict__ dinv, int n){
  int i = blockIdx.x*blockDim.x + threadIdx.x;
  if (i < n) dinv[i] = rsqrtf(fmaxf(deg[i], 1.0f));
}
__global__ void k_norme(const int* __restrict__ src, const int* __restrict__ dst,
                        const float* __restrict__ dinv, float* __restrict__ ne_, int ne){
  int e = blockIdx.x*blockDim.x + threadIdx.x;
  if (e < ne) ne_[e] = dinv[src[e]] * dinv[dst[e]];
}
// m[dst] += h[src]*norm_e  : 32 lanes per edge, 4 dims/lane (float4 gather)
__global__ void k_scatter(const float* __restrict__ h, const float* __restrict__ ne_,
                          const int* __restrict__ src, const int* __restrict__ dst,
                          float* __restrict__ m, int ne){
  long long idx = (long long)blockIdx.x*blockDim.x + threadIdx.x;
  int e = (int)(idx >> 5);
  if (e >= ne) return;
  int lane = (int)(idx & 31);
  int s = src[e], dd = dst[e];
  float w = ne_[e];
  float4 hv = *reinterpret_cast<const float4*>(h + (size_t)s*DIM + lane*4);
  float* mp = m + (size_t)dd*DIM + lane*4;
  atomicAdd(mp+0, w*hv.x); atomicAdd(mp+1, w*hv.y);
  atomicAdd(mp+2, w*hv.z); atomicAdd(mp+3, w*hv.w);
}
// h = 0.1*x + 0.9*m ; out (=|+=) hop_w[n][k] * h
__global__ void k_update(const float* __restrict__ x, const float* __restrict__ m,
                         const float* __restrict__ hw, int k,
                         float* __restrict__ h, float* __restrict__ out, long long n128){
  long long i = (long long)blockIdx.x*blockDim.x + threadIdx.x;
  if (i >= n128) return;
  int n = (int)(i >> 7);
  float hn = 0.1f*x[i] + 0.9f*m[i];
  h[i] = hn;
  float w = hw[(size_t)n*3 + k];
  if (k == 0) out[i] = w*hn; else out[i] += w*hn;
}
// hop_w = softmax(hidden @ w2 + b2) per node (hidden already ReLU'ed by GEMM epilogue)
__global__ void k_gate(const float* __restrict__ hid, const float* __restrict__ w2,
                       const float* __restrict__ b2, float* __restrict__ hw, int n){
  int i = blockIdx.x*blockDim.x + threadIdx.x;
  if (i >= n) return;
  const float* h = hid + (size_t)i*DIM;
  float l0=b2[0], l1=b2[1], l2=b2[2];
  for (int kk = 0; kk < DIM; kk++){
    float hv = h[kk];
    l0 = fmaf(hv, w2[kk*3+0], l0);
    l1 = fmaf(hv, w2[kk*3+1], l1);
    l2 = fmaf(hv, w2[kk*3+2], l2);
  }
  float mx = fmaxf(l0, fmaxf(l1, l2));
  float e0=__expf(l0-mx), e1=__expf(l1-mx), e2=__expf(l2-mx);
  float s = e0+e1+e2;
  hw[(size_t)i*3+0]=e0/s; hw[(size_t)i*3+1]=e1/s; hw[(size_t)i*3+2]=e2/s;
}

__device__ __forceinline__ float ent3(float p0, float p1, float p2){
  p0=fmaxf(p0,1e-9f); p1=fmaxf(p1,1e-9f); p2=fmaxf(p2,1e-9f);
  float e = -(p0*__logf(p0) + p1*__logf(p1) + p2*__logf(p2));
  return e / (__logf(3.0f) + 1e-9f);
}

// ---------------- per-node fusion: conflicts + 387->128->3 MLP + mix ----------------
__global__ void __launch_bounds__(128)
k_fusion(const float* __restrict__ eH, const float* __restrict__ tH, const float* __restrict__ vH,
         const float* __restrict__ w1, const float* __restrict__ b1,
         const float* __restrict__ w2, const float* __restrict__ b2,
         const float* __restrict__ csP,
         const float* __restrict__ hwe, const float* __restrict__ hwt, const float* __restrict__ hwv,
         float* __restrict__ comb, float* __restrict__ unc)
{
  int n = blockIdx.x, t = threadIdx.x;
  __shared__ float se[DIM], st_[DIM], sv_[DIM];
  __shared__ float red[6][DIM];
  __shared__ float fin[392];
  __shared__ float hid[DIM];
  __shared__ float mw[4];
  size_t base = (size_t)n*DIM;
  float e = eH[base+t], tt = tH[base+t], vv = vH[base+t];
  se[t]=e; st_[t]=tt; sv_[t]=vv;
  red[0][t]=e*e; red[1][t]=tt*tt; red[2][t]=vv*vv;
  red[3][t]=e*tt; red[4][t]=e*vv; red[5][t]=tt*vv;
  __syncthreads();
  for (int s = 64; s > 0; s >>= 1){
    if (t < s){ for (int j = 0; j < 6; j++) red[j][t] += red[j][t+s]; }
    __syncthreads();
  }
  float ne = fmaxf(sqrtf(red[0][0]),1e-8f), nt = fmaxf(sqrtf(red[1][0]),1e-8f),
        nv = fmaxf(sqrtf(red[2][0]),1e-8f);
  float c_et = 1.0f - red[3][0]/(ne*nt);
  float c_ev = 1.0f - red[4][0]/(ne*nv);
  float c_tv = 1.0f - red[5][0]/(nt*nv);
  fin[t]=e; fin[128+t]=tt; fin[256+t]=vv;
  if (t == 0){ fin[384]=c_et; fin[385]=c_ev; fin[386]=c_tv; }
  __syncthreads();
  float acc = b1[t];
  for (int r = 0; r < 387; r++) acc = fmaf(fin[r], w1[r*DIM + t], acc);
  hid[t] = fmaxf(acc, 0.0f);
  __syncthreads();
  if (t < 3){
    float lg = b2[t];
    for (int kk = 0; kk < DIM; kk++) lg = fmaf(hid[kk], w2[kk*3+t], lg);
    if (t >= 1) lg -= csP[0]*c_tv;
    mw[t] = lg;
  }
  __syncthreads();
  if (t == 0){
    float mx = fmaxf(mw[0], fmaxf(mw[1], mw[2]));
    float e0=__expf(mw[0]-mx), e1=__expf(mw[1]-mx), e2=__expf(mw[2]-mx);
    float s = e0+e1+e2;
    mw[0]=e0/s; mw[1]=e1/s; mw[2]=e2/s;
  }
  __syncthreads();
  comb[base+t] = mw[0]*se[t] + mw[1]*st_[t] + mw[2]*sv_[t];
  if (t == 0 && n >= UCNT){
    float fu = ent3(mw[0], mw[1], mw[2]);
    size_t o = (size_t)n*3;
    float hu = (ent3(hwe[o],hwe[o+1],hwe[o+2]) + ent3(hwt[o],hwt[o+1],hwt[o+2])
              + ent3(hwv[o],hwv[o+1],hwv[o+2])) * (1.0f/3.0f);
    unc[n - UCNT] = 0.5f*fu + 0.5f*hu;
  }
}

// ---------------- scoring ----------------
__global__ void k_center(const float* __restrict__ comb, float* __restrict__ center){
  int b = blockIdx.x, t = threadIdx.x;        // 128 threads = dims; 64 items/block
  float s = 0.0f;
  int i0 = b*64;
  for (int i = i0; i < i0+64; i++) s += comb[(size_t)(UCNT+i)*DIM + t];
  atomicAdd(&center[t], s);
}
__global__ void k_mm_init(unsigned* mm){ if (threadIdx.x == 0){ mm[0] = 0xFFFFFFFFu; mm[1] = 0u; } }
__global__ void k_minmax(const float* __restrict__ deg, unsigned* __restrict__ mm, int ni){
  int i = blockIdx.x*blockDim.x + threadIdx.x;
  if (i >= ni) return;
  unsigned u = __float_as_uint(deg[UCNT + i]);  // non-negative floats: uint order == float order
  atomicMin(&mm[0], u); atomicMax(&mm[1], u);
}
__global__ void k_score(const float* __restrict__ comb, const float* __restrict__ center,
                        const float* __restrict__ unc, const float* __restrict__ deg,
                        const unsigned* __restrict__ mm,
                        const float* __restrict__ impw, const float* __restrict__ impbP,
                        const float* __restrict__ coefP, float* __restrict__ score, int ni){
  int i = blockIdx.x*blockDim.x + threadIdx.x;
  if (i >= ni) return;
  const float* h = comb + (size_t)(UCNT + i)*DIM;
  const float invI = 1.0f/(float)ICNT;
  float dot=0.f, na=0.f, nb=0.f, imp=0.f;
  for (int d = 0; d < DIM; d++){
    float a = h[d], c = center[d]*invI;
    dot = fmaf(a, c, dot); na = fmaf(a, a, na); nb = fmaf(c, c, nb);
    imp = fmaf(a, impw[d], imp);
  }
  float rep = dot / (fmaxf(sqrtf(na),1e-8f)*fmaxf(sqrtf(nb),1e-8f));
  float mn = __uint_as_float(mm[0]), mx = __uint_as_float(mm[1]);
  float idg = (deg[UCNT+i] - mn) / (mx - mn + 1e-9f);
  float c0 = log1pf(__expf(coefP[0])), c1 = log1pf(__expf(coefP[1]));
  float c2 = log1pf(__expf(coefP[2])), c3 = log1pf(__expf(coefP[3]));
  score[i] = c0*rep + c1*unc[i] + c2*idg + c3*(imp + impbP[0]);
}
__global__ void k_top7(const float* __restrict__ score, int n, int* __restrict__ topidx){
  __shared__ float sv[256]; __shared__ int si[256]; __shared__ int chosen[8];
  int t = threadIdx.x;
  for (int p = 0; p < 7; p++){
    float best = -3.4e38f; int bi = n;
    for (int i = t; i < n; i += 256){
      bool skip = false;
      for (int q = 0; q < p; q++) if (chosen[q] == i) skip = true;
      if (skip) continue;
      float v = score[i];
      if (v > best || (v == best && i < bi)){ best = v; bi = i; }
    }
    sv[t] = best; si[t] = bi; __syncthreads();
    for (int s = 128; s > 0; s >>= 1){
      if (t < s){
        if (sv[t+s] > sv[t] || (sv[t+s] == sv[t] && si[t+s] < si[t])){ sv[t]=sv[t+s]; si[t]=si[t+s]; }
      }
      __syncthreads();
    }
    if (t == 0){ chosen[p] = si[0]; topidx[p] = si[0]; }
    __syncthreads();
  }
}
// z_mem: users -> 0 ; items -> slot0 = item_h, slots1..7 = item_h[top_idx[r-1]]
__global__ void k_zmem(const float* __restrict__ comb, const int* __restrict__ topidx,
                       float* __restrict__ z){
  long long idx = (long long)blockIdx.x*blockDim.x + threadIdx.x;
  if (idx >= (long long)NNODE*8*DIM) return;
  int n = (int)(idx >> 10);
  int rem = (int)(idx & 1023);
  int r = rem >> 7, d = rem & 127;
  float v = 0.0f;
  if (n >= UCNT){
    int src = (r == 0) ? n : (UCNT + topidx[r-1]);
    v = comb[(size_t)src*DIM + d];
  }
  z[idx] = v;
}

// ---------------- host ----------------
static inline unsigned nblk(long long n, int b){ return (unsigned)((n + b - 1)/b); }

extern "C" void kernel_launch(void* const* d_in, const int* in_sizes, int n_in,
                              void* d_out, int out_size, void* d_ws, size_t ws_size,
                              hipStream_t stream) {
  (void)in_sizes; (void)n_in; (void)out_size; (void)ws_size;
  const float* user_emb = (const float*)d_in[0];
  const float* item_v   = (const float*)d_in[1];
  const float* item_t   = (const float*)d_in[2];
  const float* t_w1=(const float*)d_in[3],  *t_b1=(const float*)d_in[4],  *t_g1=(const float*)d_in[5],
             * t_bt1=(const float*)d_in[6], *t_a1=(const float*)d_in[7];
  const float* t_w2=(const float*)d_in[8],  *t_b2=(const float*)d_in[9],  *t_g2=(const float*)d_in[10],
             * t_bt2=(const float*)d_in[11],*t_a2=(const float*)d_in[12];
  const float* v_w1=(const float*)d_in[13], *v_b1=(const float*)d_in[14], *v_g1=(const float*)d_in[15],
             * v_bt1=(const float*)d_in[16],*v_a1=(const float*)d_in[17];
  const float* v_w2=(const float*)d_in[18], *v_b2=(const float*)d_in[19], *v_g2=(const float*)d_in[20],
             * v_bt2=(const float*)d_in[21],*v_a2=(const float*)d_in[22];
  const float* hg_w1[3] = {(const float*)d_in[23],(const float*)d_in[27],(const float*)d_in[31]};
  const float* hg_b1[3] = {(const float*)d_in[24],(const float*)d_in[28],(const float*)d_in[32]};
  const float* hg_w2[3] = {(const float*)d_in[25],(const float*)d_in[29],(const float*)d_in[33]};
  const float* hg_b2[3] = {(const float*)d_in[26],(const float*)d_in[30],(const float*)d_in[34]};
  const float* fus_w1=(const float*)d_in[35], *fus_b1=(const float*)d_in[36];
  const float* fus_w2=(const float*)d_in[37], *fus_b2=(const float*)d_in[38];
  const float* cscale=(const float*)d_in[39];
  const float* imp_w=(const float*)d_in[40], *imp_b=(const float*)d_in[41];
  const float* coef =(const float*)d_in[42];
  const int* esrc = (const int*)d_in[43];
  const int* edst = (const int*)d_in[44];

  // ---- workspace carve ----
  char* w = (char*)d_ws;
  auto take = [&](size_t bytes)->void*{ void* p=(void*)w; w += (bytes + 255) & ~(size_t)255; return p; };
  __bf16* bt_tw1 = (__bf16*)take((size_t)HID1*TIN*2);
  __bf16* bt_vw1 = (__bf16*)take((size_t)HID1*VIN*2);
  __bf16* bt_tw2 = (__bf16*)take((size_t)DIM*HID1*2);
  __bf16* bt_vw2 = (__bf16*)take((size_t)DIM*HID1*2);
  __bf16* bt_hg[3];
  for (int m = 0; m < 3; m++) bt_hg[m] = (__bf16*)take((size_t)DIM*DIM*2);
  float* h1    = (float*)take((size_t)ICNT*HID1*4);
  float* x_e   = (float*)take((size_t)NNODE*DIM*4);
  float* x_t   = (float*)take((size_t)NNODE*DIM*4);
  float* x_v   = (float*)take((size_t)NNODE*DIM*4);
  float* out_e = (float*)take((size_t)NNODE*DIM*4);
  float* out_t = (float*)take((size_t)NNODE*DIM*4);
  float* out_v = (float*)take((size_t)NNODE*DIM*4);
  float* hbuf  = (float*)take((size_t)NNODE*DIM*4);
  float* mbuf  = (float*)take((size_t)NNODE*DIM*4);
  float* hw_e  = (float*)take((size_t)NNODE*3*4);
  float* hw_t  = (float*)take((size_t)NNODE*3*4);
  float* hw_v  = (float*)take((size_t)NNODE*3*4);
  float* deg   = (float*)take((size_t)NNODE*4);
  float* dinv  = (float*)take((size_t)NNODE*4);
  float* norme = (float*)take((size_t)E2CNT*4);
  float* unc   = (float*)take((size_t)ICNT*4);
  float* center= (float*)take(DIM*4);
  unsigned* mm = (unsigned*)take(64);
  int* topidx  = (int*)take(64);

  float* comb  = (float*)d_out;                                  // [NNODE,128]
  float* zmem  = comb + (size_t)NNODE*DIM;                       // [NNODE,8,128]
  float* score = zmem + (size_t)NNODE*8*DIM;                     // [ICNT]

  const int B = 256;
  // ---- 1) weight transpose/convert to bf16 [N][K] ----
  k_conv_bt<<<nblk((long long)TIN*HID1,B),B,0,stream>>>(t_w1, bt_tw1, TIN, HID1);
  k_conv_bt<<<nblk((long long)VIN*HID1,B),B,0,stream>>>(v_w1, bt_vw1, VIN, HID1);
  k_conv_bt<<<nblk((long long)HID1*DIM,B),B,0,stream>>>(t_w2, bt_tw2, HID1, DIM);
  k_conv_bt<<<nblk((long long)HID1*DIM,B),B,0,stream>>>(v_w2, bt_vw2, HID1, DIM);
  for (int m = 0; m < 3; m++)
    k_conv_bt<<<nblk((long long)DIM*DIM,B),B,0,stream>>>(hg_w1[m], bt_hg[m], DIM, DIM);

  // ---- 2) item encoders (TDM-staged WMMA GEMM + fused BN/PReLU epilogue) ----
  k_gemm_bf16<<<dim3(ICNT/16, HID1/64),128,0,stream>>>(item_t, TIN,  bt_tw1, h1, HID1,
                                                       t_b1, t_g1, t_bt1, t_a1, 1);
  k_gemm_bf16<<<dim3(ICNT/16, DIM/64),128,0,stream>>>(h1, HID1, bt_tw2, x_t + (size_t)UCNT*DIM, DIM,
                                                       t_b2, t_g2, t_bt2, t_a2, 1);
  k_gemm_bf16<<<dim3(ICNT/16, HID1/64),128,0,stream>>>(item_v, VIN,  bt_vw1, h1, HID1,
                                                       v_b1, v_g1, v_bt1, v_a1, 1);
  k_gemm_bf16<<<dim3(ICNT/16, DIM/64),128,0,stream>>>(h1, HID1, bt_vw2, x_v + (size_t)UCNT*DIM, DIM,
                                                       v_b2, v_g2, v_bt2, v_a2, 1);

  // ---- 3) node inputs: x_e = [emb;0], x_t = [0;enc_t], x_v = [0;enc_v] ----
  k_copy<<<nblk((long long)UCNT*DIM,B),B,0,stream>>>(x_e, user_emb, (long long)UCNT*DIM);
  k_zero<<<nblk((long long)ICNT*DIM,B),B,0,stream>>>(x_e + (size_t)UCNT*DIM, (long long)ICNT*DIM);
  k_zero<<<nblk((long long)UCNT*DIM,B),B,0,stream>>>(x_t, (long long)UCNT*DIM);
  k_zero<<<nblk((long long)UCNT*DIM,B),B,0,stream>>>(x_v, (long long)UCNT*DIM);

  // ---- 4) degrees / edge norm ----
  k_zero<<<nblk(NNODE,B),B,0,stream>>>(deg, NNODE);
  k_deg<<<nblk(E2CNT,B),B,0,stream>>>(edst, deg, E2CNT);
  k_dinv<<<nblk(NNODE,B),B,0,stream>>>(deg, dinv, NNODE);
  k_norme<<<nblk(E2CNT,B),B,0,stream>>>(esrc, edst, dinv, norme, E2CNT);

  // ---- 5) soft-hop per modality ----
  const float* xs[3]  = {x_e, x_t, x_v};
  float* outs[3]      = {out_e, out_t, out_v};
  float* hws[3]       = {hw_e, hw_t, hw_v};
  const long long ND = (long long)NNODE*DIM;
  for (int m = 0; m < 3; m++){
    // hop gate: hidden = relu(x@w1+b1) via WMMA, then softmax(hidden@w2+b2)
    k_gemm_bf16<<<dim3(NNODE/16, DIM/64),128,0,stream>>>(xs[m], DIM, bt_hg[m], mbuf, DIM,
                                                         hg_b1[m], nullptr, nullptr, nullptr, 2);
    k_gate<<<nblk(NNODE,B),B,0,stream>>>(mbuf, hg_w2[m], hg_b2[m], hws[m], NNODE);
    k_copy<<<nblk(ND,B),B,0,stream>>>(hbuf, xs[m], ND);
    for (int k = 0; k < 3; k++){
      k_zero<<<nblk(ND,B),B,0,stream>>>(mbuf, ND);
      k_scatter<<<nblk((long long)E2CNT*32,B),B,0,stream>>>(hbuf, norme, esrc, edst, mbuf, E2CNT);
      k_update<<<nblk(ND,B),B,0,stream>>>(xs[m], mbuf, hws[m], k, hbuf, outs[m], ND);
    }
  }

  // ---- 6) fusion -> combined (d_out), uncertainty for items ----
  k_fusion<<<NNODE,128,0,stream>>>(out_e, out_t, out_v, fus_w1, fus_b1, fus_w2, fus_b2,
                                   cscale, hw_e, hw_t, hw_v, comb, unc);

  // ---- 7) item scoring ----
  k_zero<<<1,DIM,0,stream>>>(center, DIM);
  k_center<<<ICNT/64,128,0,stream>>>(comb, center);
  k_mm_init<<<1,32,0,stream>>>(mm);
  k_minmax<<<nblk(ICNT,B),B,0,stream>>>(deg, mm, ICNT);
  k_score<<<nblk(ICNT,B),B,0,stream>>>(comb, center, unc, deg, mm, imp_w, imp_b, coef, score, ICNT);
  k_top7<<<1,256,0,stream>>>(score, ICNT, topidx);

  // ---- 8) z_mem assembly ----
  k_zmem<<<nblk((long long)NNODE*8*DIM,B),B,0,stream>>>(comb, topidx, zmem);
}
